// NBT_model_72791105733294
// MI455X (gfx1250) — matrix operand order repaired and checked by reading, not compiled, not run on hardware
//
#include <hip/hip_runtime.h>
#include <cmath>

// Problem dims (fixed by reference)
#define B_  128
#define C_  1000
#define D_  300
#define L_  40
#define H_  100
#define KP  320      // K (=D) padded to multiple of 32 for f16 WMMA
#define KT_ (KP/32)  // 10 WMMA K-steps
#define CT  63       // ceil(C/16)
#define NT  7        // ceil(H/16)

typedef __attribute__((ext_vector_type(16))) _Float16 v16h;
typedef __attribute__((ext_vector_type(8)))  _Float16 v8h;
typedef __attribute__((ext_vector_type(8)))  float    v8f;

__device__ __forceinline__ float sigf(float x) { return 1.0f / (1.0f + __expf(-x)); }

// ---------------------------------------------------------------------------
// Kernel 1: CNN utterance encoder.  One block per batch row b.
// fu[b,f] = sum_k relu( max_t conv_k[b,f,t] + b_k[f] )
// ---------------------------------------------------------------------------
__global__ void __launch_bounds__(256)
k_conv_encoder(const float* __restrict__ utt,
               const float* __restrict__ w1, const float* __restrict__ b1,
               const float* __restrict__ w2, const float* __restrict__ b2,
               const float* __restrict__ w3, const float* __restrict__ b3,
               float* __restrict__ fu) {
  __shared__ float xs[L_ * D_];                 // utt[b] as [L][D], 48 KB
  const int b   = blockIdx.x;
  const int tid = threadIdx.x;
  for (int i = tid; i < L_ * D_; i += 256) xs[i] = utt[b * L_ * D_ + i];
  __syncthreads();

  for (int f = tid; f < D_; f += 256) {
    float r1, r2, r3;
    {   // width-1 branch
      float a[L_];
      #pragma unroll
      for (int t = 0; t < L_; ++t) a[t] = 0.0f;
      const float* wf = w1 + f * D_;            // [D][1]
      for (int d = 0; d < D_; ++d) {
        const float w = wf[d];
        #pragma unroll
        for (int t = 0; t < L_; ++t) a[t] += xs[t * D_ + d] * w;
      }
      float m = a[0];
      #pragma unroll
      for (int t = 1; t < L_; ++t) m = fmaxf(m, a[t]);
      r1 = fmaxf(m + b1[f], 0.0f);
    }
    {   // width-2 branch
      float a[L_ - 1];
      #pragma unroll
      for (int t = 0; t < L_ - 1; ++t) a[t] = 0.0f;
      const float* wf = w2 + f * D_ * 2;        // [D][2]
      for (int d = 0; d < D_; ++d) {
        const float w0 = wf[d * 2 + 0], wA = wf[d * 2 + 1];
        #pragma unroll
        for (int t = 0; t < L_ - 1; ++t)
          a[t] += xs[t * D_ + d] * w0 + xs[(t + 1) * D_ + d] * wA;
      }
      float m = a[0];
      #pragma unroll
      for (int t = 1; t < L_ - 1; ++t) m = fmaxf(m, a[t]);
      r2 = fmaxf(m + b2[f], 0.0f);
    }
    {   // width-3 branch
      float a[L_ - 2];
      #pragma unroll
      for (int t = 0; t < L_ - 2; ++t) a[t] = 0.0f;
      const float* wf = w3 + f * D_ * 3;        // [D][3]
      for (int d = 0; d < D_; ++d) {
        const float w0 = wf[d * 3 + 0], wA = wf[d * 3 + 1], wB = wf[d * 3 + 2];
        #pragma unroll
        for (int t = 0; t < L_ - 2; ++t)
          a[t] += xs[t * D_ + d] * w0 + xs[(t + 1) * D_ + d] * wA
                + xs[(t + 2) * D_ + d] * wB;
      }
      float m = a[0];
      #pragma unroll
      for (int t = 1; t < L_ - 2; ++t) m = fmaxf(m, a[t]);
      r3 = fmaxf(m + b3[f], 0.0f);
    }
    fu[b * D_ + f] = r1 + r2 + r3;
  }
}

// ---------------------------------------------------------------------------
// Kernel 2: cand[c,e] = sigmoid( [slot_emb|value_emb] @ Wc.T + bc ).  1 block/c.
// ---------------------------------------------------------------------------
__global__ void __launch_bounds__(128)
k_cand(const float* __restrict__ slot_emb, const float* __restrict__ value_emb,
       const float* __restrict__ Wc, const float* __restrict__ bc,
       float* __restrict__ cand) {
  __shared__ float se[D_], ve[D_];
  const int c = blockIdx.x, tid = threadIdx.x;
  for (int i = tid; i < D_; i += 128) { se[i] = slot_emb[c * D_ + i]; ve[i] = value_emb[c * D_ + i]; }
  __syncthreads();
  for (int e = tid; e < D_; e += 128) {
    const float* wr = Wc + e * (2 * D_);
    float acc = bc[e];
    for (int d = 0; d < D_; ++d) acc += se[d] * wr[d] + ve[d] * wr[D_ + d];
    cand[c * D_ + e] = sigf(acc);
  }
}

// ---------------------------------------------------------------------------
// Kernel 3: batch-independent gates.  y23[c] = y2[c] + y3[c].  1 block/c.
// ---------------------------------------------------------------------------
__global__ void __launch_bounds__(128)
k_gates(const float* __restrict__ slot_emb, const float* __restrict__ value_emb,
        const float* __restrict__ sys_slots, const float* __restrict__ conf_slots,
        const float* __restrict__ conf_values, const float* __restrict__ fu,
        const float* __restrict__ Wmr, const float* __restrict__ bmr,
        const float* __restrict__ Wmc, const float* __restrict__ bmc,
        const float* __restrict__ Wj,  const float* __restrict__ bj,
        float* __restrict__ y23) {
  __shared__ float se[D_], ve[D_];
  __shared__ float sarr[B_], tarr[B_];
  __shared__ float smr[D_], smc[D_];
  __shared__ float red;
  const int c = blockIdx.x, tid = threadIdx.x;
  for (int i = tid; i < D_; i += 128) { se[i] = slot_emb[c * D_ + i]; ve[i] = value_emb[c * D_ + i]; }
  if (tid == 0) red = 0.0f;
  __syncthreads();

  {   // s[b] = se.sys_slots[b];  t[b] = (se.conf_slots[b]) * (ve.conf_values[b])
    const int b = tid;
    float s1 = 0.f, s2 = 0.f, s3 = 0.f;
    const float* ss = sys_slots  + b * D_;
    const float* cs = conf_slots + b * D_;
    const float* cv = conf_values + b * D_;
    for (int d = 0; d < D_; ++d) { s1 += se[d] * ss[d]; s2 += se[d] * cs[d]; s3 += ve[d] * cv[d]; }
    sarr[b] = s1;  tarr[b] = s2 * s3;
  }
  __syncthreads();

  for (int d = tid; d < D_; d += 128) {   // m_r/m_c rows, pre-sigmoided
    float mr = 0.f, mc = 0.f;
    for (int b = 0; b < B_; ++b) {
      const float f = fu[b * D_ + d];
      mr += sarr[b] * f;  mc += tarr[b] * f;
    }
    smr[d] = sigf(mr);  smc[d] = sigf(mc);
  }
  __syncthreads();

  if (tid < H_) {                          // hidden layers + Wj contraction
    const int h = tid;
    float hr = bmr[h], hc = bmc[h];
    const float* wr = Wmr + h * D_;
    const float* wc = Wmc + h * D_;
    for (int d = 0; d < D_; ++d) { hr += smr[d] * wr[d]; hc += smc[d] * wc[d]; }
    atomicAdd(&red, (sigf(hr) + sigf(hc)) * Wj[h]);
  }
  __syncthreads();
  if (tid == 0) y23[c] = red + 2.0f * bj[0];
}

// ---------------------------------------------------------------------------
// Kernel 4: dominant GEMM via f16 WMMA with f32 accumulate.
//   hvec[b,c,h] = sum_d sigmoid(cand[c,d]*fu[b,d]) * Wd[h,d] + bd[h]
//   y1[b,c]     = sum_h sigmoid(hvec)*Wj[h] + bj
//   out[b,c]    = 0.5*(y1 + y23[c]) + 0.5*y_past[b,c]
// Block: 128 threads (4 waves); grid (CT, B).  Wave w handles N-tiles w, w+4.
// Both A and B fragments are fully register-resident before the WMMA chain so
// the 10 v_wmma issue back-to-back behind a single s_wait_dscnt.
// ---------------------------------------------------------------------------
__global__ void __launch_bounds__(128)
k_big_gemm(const float* __restrict__ cand, const float* __restrict__ fu,
           const float* __restrict__ Wd,  const float* __restrict__ bd,
           const float* __restrict__ Wj,  const float* __restrict__ bj,
           const float* __restrict__ y23, const float* __restrict__ ypast,
           float* __restrict__ out) {
  __shared__ alignas(32) _Float16 sa[16][KP];        // A tile: 10 KB
  __shared__ alignas(32) _Float16 wdt[4][16][KP];    // B tiles (per wave): 40 KB
  __shared__ float y1acc[16];

  const int c0   = blockIdx.x * 16;
  const int b    = blockIdx.y;
  const int tid  = threadIdx.x;
  const int wave = tid >> 5;
  const int lane = tid & 31;

  if (tid < 16) y1acc[tid] = 0.0f;

  // Stage A = sigmoid(cand * fu) in f16, zero-padded rows (c>=C) and cols (k>=D)
  for (int i = tid; i < 16 * KP; i += 128) {
    const int m = i / KP, k = i % KP;
    const int c = c0 + m;
    _Float16 hv = (_Float16)0.0f;
    if (c < C_ && k < D_)
      hv = (_Float16)sigf(cand[c * D_ + k] * fu[b * D_ + k]);
    sa[m][k] = hv;
  }
  __syncthreads();

  const int mn = lane & 15;   // A row (M) == B/C column (N) for this lane
  const int hi = lane >> 4;   // half-wave selector

  // Preload ALL A fragments into registers (contiguous k order, same as B):
  // lanes 0-15 hold K 0..15 +32*kt; lanes 16-31 hold K 16..31 +32*kt
  v16h av[KT_];
  #pragma unroll
  for (int kt = 0; kt < KT_; ++kt) {
    const int kb = kt * 32;
    av[kt] = *(const v16h*)&sa[mn][kb + hi * 16];
  }

  for (int nt = wave; nt < NT; nt += 4) {
    __builtin_prefetch(Wd + (nt * 16) * D_, 0, 1);
    // Stage B tile: wdt[wave][n][k] = Wd[nt*16+n][k] (zero-padded), f16
    for (int i = lane; i < 16 * KP; i += 32) {
      const int n = i / KP, k = i % KP;
      const int hcol = nt * 16 + n;
      wdt[wave][n][k] = (hcol < H_ && k < D_) ? (_Float16)Wd[hcol * D_ + k]
                                              : (_Float16)0.0f;
    }

    // Preload ALL B fragments: lanes 0-15 K 0..15, lanes 16-31 K 16..31; N = mn
    v16h bv[KT_];
    #pragma unroll
    for (int kt = 0; kt < KT_; ++kt)
      bv[kt] = *(const v16h*)&wdt[wave][mn][kt * 32 + hi * 16];

    // Back-to-back WMMA accumulator chain over K
    v8f acc = {};
    #pragma unroll
    for (int kt = 0; kt < KT_; ++kt)
      acc = __builtin_amdgcn_wmma_f32_16x16x32_f16(
          /*neg_a=*/false, av[kt], /*neg_b=*/false, bv[kt],
          /*c_mod=*/(short)0, acc, /*reuse_a=*/false, /*reuse_b=*/false);

    // Epilogue: C layout — VGPR r, lanes 0-15: M=r; lanes 16-31: M=r+8; N=mn
    const int hcol = nt * 16 + mn;
    if (hcol < H_) {
      const float bdv = bd[hcol], wjv = Wj[hcol];
      #pragma unroll
      for (int r = 0; r < 8; ++r)
        atomicAdd(&y1acc[r + hi * 8], sigf(acc[r] + bdv) * wjv);
    }
  }
  __syncthreads();

  if (tid < 16) {
    const int c = c0 + tid;
    if (c < C_) {
      const float y1 = y1acc[tid] + bj[0];
      out[b * C_ + c] = 0.5f * (y1 + y23[c]) + 0.5f * ypast[b * C_ + c];
    }
  }
}

// ---------------------------------------------------------------------------
extern "C" void kernel_launch(void* const* d_in, const int* in_sizes, int n_in,
                              void* d_out, int out_size, void* d_ws, size_t ws_size,
                              hipStream_t stream) {
  (void)in_sizes; (void)n_in; (void)out_size; (void)ws_size;
  const float* utt         = (const float*)d_in[0];
  const float* sys_slots   = (const float*)d_in[1];
  const float* conf_slots  = (const float*)d_in[2];
  const float* conf_values = (const float*)d_in[3];
  const float* ypast       = (const float*)d_in[4];
  const float* slot_emb    = (const float*)d_in[5];
  const float* value_emb   = (const float*)d_in[6];
  const float* Wc          = (const float*)d_in[7];
  const float* bc          = (const float*)d_in[8];
  const float* w1          = (const float*)d_in[9];
  const float* b1          = (const float*)d_in[10];
  const float* w2          = (const float*)d_in[11];
  const float* b2          = (const float*)d_in[12];
  const float* w3          = (const float*)d_in[13];
  const float* b3          = (const float*)d_in[14];
  const float* Wd          = (const float*)d_in[15];
  const float* bd          = (const float*)d_in[16];
  const float* Wmr         = (const float*)d_in[17];
  const float* bmr         = (const float*)d_in[18];
  const float* Wmc         = (const float*)d_in[19];
  const float* bmc         = (const float*)d_in[20];
  const float* Wj          = (const float*)d_in[21];
  const float* bj          = (const float*)d_in[22];

  float* out  = (float*)d_out;
  float* ws   = (float*)d_ws;
  float* fu   = ws;                    // B*D   = 38400 floats
  float* cand = fu + B_ * D_;          // C*D   = 300000 floats
  float* y23  = cand + C_ * D_;        // C     = 1000 floats

  k_conv_encoder<<<B_, 256, 0, stream>>>(utt, w1, b1, w2, b2, w3, b3, fu);
  k_cand<<<C_, 128, 0, stream>>>(slot_emb, value_emb, Wc, bc, cand);
  k_gates<<<C_, 128, 0, stream>>>(slot_emb, value_emb, sys_slots, conf_slots,
                                  conf_values, fu, Wmr, bmr, Wmc, bmc, Wj, bj, y23);
  dim3 g4(CT, B_);
  k_big_gemm<<<g4, 128, 0, stream>>>(cand, fu, Wd, bd, Wj, bj, y23, ypast, out);
}
